// HolohoverModelBlack_31387620999405
// MI455X (gfx1250) — matrix operand order repaired and checked by reading, not compiled, not run on hardware
//
#include <hip/hip_runtime.h>

typedef _Float16 half_t;
typedef _Float16 v16h __attribute__((ext_vector_type(16)));
typedef float    v8f  __attribute__((ext_vector_type(8)));

#define WAVES 4
#define ACTW  224   // activation arena width (halves), covers K up to 224 (FNN3)
#define SIGW  160   // s1[0..63] s2[64..127] s3[128..159]
#define ROWW  56    // f32 row scratch width
#define RX  0
#define RF  6
#define RDY 12
#define RG  18
#define RY  54

// fragment table (each fragment = 32 lanes * 16 halves = 512 halves = 1KB)
enum {
  F_FNN1 = 0,    // 5   (ks=1, nt=5)   B = fnn_w1^T
  F_FNN2 = 5,    // 39  (ks=3, nt=13)  B = fnn_w2^T
  F_FNN3 = 44,   // 7   (ks=7, nt=1)   B = fnn_w3^T
  F_GNN  = 51,   // 3                  B = gnn_w^T
  F_IW1  = 54,   // 4                  B = icnn_w1^T
  F_IW2  = 58,   // 8   (ks=2, nt=4)   B = icnn_w2^T
  F_IM2  = 66,   // 4                  B = icnn_im2^T
  F_IW3  = 70,   // 4   (ks=2, nt=2)   B = icnn_w3^T
  F_IM3  = 74,   // 2                  B = icnn_im3^T
  F_BW3  = 76,   // 4   (ks=1, nt=4)   B = icnn_w3   (backward)
  F_BW2  = 80,   // 8   (ks=2, nt=4)   B = icnn_w2   (backward)
  F_BW1  = 88,   // 2   (ks=2, nt=1)   B = icnn_w1   (backward)
  F_BI2  = 90,   // 2   (ks=2, nt=1)   B = icnn_im2  (backward)
  F_BI3  = 92,   // 1   (ks=1, nt=1)   B = icnn_im3  (backward)
  F_TOTAL = 93
};
#define YREF_BYTE_OFF (F_TOTAL * 512 * 2)

#define LOG2E_F 1.4426950408889634f
#define LN2_F   0.6931471805599453f

// ---- gfx1250 hardware transcendentals, branchless ----
__device__ inline float fast_tanh(float x){
#if __has_builtin(__builtin_amdgcn_tanhf)
  return __builtin_amdgcn_tanhf(x);            // v_tanh_f32
#else
  float r;
  asm("v_tanh_f32 %0, %1\n\tv_nop\n\tv_nop" : "=v"(r) : "v"(x));
  return r;
#endif
}
__device__ inline float fast_exp2(float x){ return __builtin_amdgcn_exp2f(x); }  // v_exp_f32
__device__ inline float fast_log2(float x){ return __builtin_amdgcn_logf(x);  }  // v_log_f32
__device__ inline float fast_rcp (float x){ return __builtin_amdgcn_rcpf(x);  }  // v_rcp_f32

// sigmoid(z) = 1 / (1 + 2^(-z*log2e))
__device__ inline float sigmoid_f(float z){
  return fast_rcp(1.f + fast_exp2(-z * LOG2E_F));
}
// softplus(z) = max(z,0) + ln2 * log2(1 + 2^(-|z|*log2e))
__device__ inline float softplus_f(float z){
  float t = fast_exp2(-fabsf(z) * LOG2E_F);
  return fmaxf(z, 0.f) + LN2_F * fast_log2(1.f + t);
}

__device__ inline v16h frag_ld(const half_t* __restrict__ frag, int f, int lane){
  return ((const v16h*)(frag + (size_t)f * 512))[lane];
}

// Gather a 16x32 f16 A-fragment from row-major LDS activations.
// ISA layout: lanes 0-15 row M=lane, lanes16-31 same rows upper-K half;
// VGPR v (pair p): K = (v<4?0:16) + (lane>=16?8:0) + (v%4)*2 + p
__device__ inline v16h lds_ldA(const half_t* __restrict__ act, int lane, int kbase){
  const int r = lane & 15, hf = lane >> 4;
  const half_t* row = act + r * ACTW + kbase;
  v16h a;
#pragma unroll
  for (int v = 0; v < 8; ++v){
    int k = ((v & 4) << 2) + hf * 8 + ((v & 3) << 1);
    a[2*v]   = row[k];
    a[2*v+1] = row[k + 1];
  }
  return a;
}

__device__ inline v8f wmma_f16(v16h a, v16h b, v8f c){
  return __builtin_amdgcn_wmma_f32_16x16x32_f16(false, a, false, b, (short)0, c, false, false);
}

// ---- weight prepack into B-fragment layout: frag[lane*16+i] = B[kbase + (i<8 ? 8*(lane>>4)+i : 16+8*(lane>>4)+i-8)][nbase + (lane&15)]
// transposed=1: B = W^T  (K=indim, N=outdim);  transposed=0: B = W (K=outdim, N=indim)
__global__ void holo_prepack(const float* __restrict__ W, int outdim, int indim,
                             int transposed, half_t* __restrict__ dst, int ksteps)
{
  int bid = blockIdx.x;
  int kt = bid % ksteps, nt = bid / ksteps;
  int t = threadIdx.x;           // 0..511
  int L = t >> 4, i = t & 15;
  int k = kt * 32 + ((i < 8) ? (8 * (L >> 4) + i) : (16 + 8 * (L >> 4) + (i - 8)));
  int n = nt * 16 + (L & 15);
  float v;
  if (transposed) v = (k < indim  && n < outdim) ? W[n * indim + k] : 0.f;
  else            v = (k < outdim && n < indim ) ? W[k * indim + n] : 0.f;
  dst[(size_t)bid * 512 + t] = (half_t)v;
}

// ---- icnn(xref) scalar, f32, one lane
__global__ void holo_yref(const float* __restrict__ xref,
    const float* __restrict__ w1, const float* __restrict__ b1,
    const float* __restrict__ w2, const float* __restrict__ b2,
    const float* __restrict__ w3, const float* __restrict__ b3,
    const float* __restrict__ w4, const float* __restrict__ b4,
    const float* __restrict__ im2, const float* __restrict__ im3,
    const float* __restrict__ im4, float* __restrict__ yref)
{
  if (threadIdx.x != 0 || blockIdx.x != 0) return;
  float h1[60], h2[60], h3[30];
  for (int i = 0; i < 60; ++i){
    float z = b1[i];
    for (int d = 0; d < 6; ++d) z += w1[i*6+d] * xref[d];
    h1[i] = softplus_f(z);
  }
  for (int i = 0; i < 60; ++i){
    float z = b2[i];
    for (int j = 0; j < 60; ++j) z += w2[i*60+j] * h1[j];
    for (int d = 0; d < 6; ++d)  z += im2[i*6+d] * xref[d];
    h2[i] = softplus_f(z);
  }
  for (int i = 0; i < 30; ++i){
    float z = b3[i];
    for (int j = 0; j < 60; ++j) z += w3[i*60+j] * h2[j];
    for (int d = 0; d < 6; ++d)  z += im3[i*6+d] * xref[d];
    h3[i] = softplus_f(z);
  }
  float y = b4[0];
  for (int j = 0; j < 30; ++j) y += w4[j] * h3[j];
  for (int d = 0; d < 6; ++d)  y += im4[d] * xref[d];
  *yref = y;
}

// ---- main: one wave = 16 samples; all matmuls via v_wmma_f32_16x16x32_f16
__global__ __launch_bounds__(32 * WAVES) void holo_main(
    const float* __restrict__ X, const float* __restrict__ U,
    const float* __restrict__ xref,
    const float* __restrict__ fb1, const float* __restrict__ fb2, const float* __restrict__ fb3,
    const float* __restrict__ gb,
    const float* __restrict__ ib1, const float* __restrict__ ib2, const float* __restrict__ ib3,
    const float* __restrict__ ib4,
    const float* __restrict__ w4, const float* __restrict__ im4,
    const half_t* __restrict__ frag, const float* __restrict__ yref_p,
    float* __restrict__ out)
{
  __shared__ half_t s_act[WAVES][16][ACTW];
  __shared__ half_t s_sig[WAVES][16][SIGW];
  __shared__ float  s_row[WAVES][16][ROWW];

  const int lane = threadIdx.x & 31;
  const int w    = threadIdx.x >> 5;
  const int r16  = lane & 15;
  const int hf   = lane >> 4;
  const int rowbase = (blockIdx.x * WAVES + w) * 16;

  half_t* act = &s_act[w][0][0];
  half_t* sg  = &s_sig[w][0][0];
  float*  rb  = &s_row[w][0][0];
  const v8f zacc = {0.f,0.f,0.f,0.f,0.f,0.f,0.f,0.f};

  // zero wave-private arenas (guarantees finite zero padding everywhere)
  for (int i = lane; i < 16*ACTW; i += 32) act[i] = (half_t)0.f;
  for (int i = lane; i < 16*SIGW; i += 32) sg[i]  = (half_t)0.f;

  // load 16x6 X tile (coalesced) into f32 row scratch
  for (int t = lane; t < 96; t += 32)
    rb[(t/6)*ROWW + RX + (t%6)] = X[rowbase*6 + t];
  __builtin_amdgcn_wave_barrier();

  // A_X fragment (K=6 padded to 32), reused by FNN1 / GNN / ICNN z1,z2,z3 skip terms
  v16h aX;
  {
    const float* xr = rb + r16*ROWW + RX;
#pragma unroll
    for (int v = 0; v < 8; ++v){
      int k = ((v & 4) << 2) + hf*8 + ((v & 3) << 1);
      aX[2*v]   = (half_t)((k   < 6) ? xr[k]   : 0.f);
      aX[2*v+1] = (half_t)((k+1 < 6) ? xr[k+1] : 0.f);
    }
  }

  // ---------------- FNN layer 1: 6 -> 80, tanh ----------------
#pragma unroll
  for (int nt = 0; nt < 5; ++nt){
    v8f c = zacc;
    c = wmma_f16(aX, frag_ld(frag, F_FNN1 + nt, lane), c);
    int n = nt*16 + r16;
    float bv = fb1[n];
#pragma unroll
    for (int rv = 0; rv < 8; ++rv)
      act[(rv + 8*hf)*ACTW + n] = (half_t)fast_tanh(c[rv] + bv);
  }
  __builtin_amdgcn_wave_barrier();

  // ---------------- FNN layer 2: 80 -> 200, tanh ----------------
  {
    v16h a0 = lds_ldA(act, lane, 0);
    v16h a1 = lds_ldA(act, lane, 32);
    v16h a2 = lds_ldA(act, lane, 64);
    __builtin_amdgcn_wave_barrier();
    for (int nt = 0; nt < 13; ++nt){
      v8f c = zacc;
      c = wmma_f16(a0, frag_ld(frag, F_FNN2 + nt*3 + 0, lane), c);
      c = wmma_f16(a1, frag_ld(frag, F_FNN2 + nt*3 + 1, lane), c);
      c = wmma_f16(a2, frag_ld(frag, F_FNN2 + nt*3 + 2, lane), c);
      int n = nt*16 + r16;
      float bv = (n < 200) ? fb2[n] : 0.f;
#pragma unroll
      for (int rv = 0; rv < 8; ++rv)
        act[(rv + 8*hf)*ACTW + n] = (half_t)fast_tanh(c[rv] + bv);
    }
  }
  __builtin_amdgcn_wave_barrier();

  // ---------------- FNN layer 3: 200 -> 6 (f_X) ----------------
  {
    v16h af[7];
#pragma unroll
    for (int kt = 0; kt < 7; ++kt) af[kt] = lds_ldA(act, lane, kt*32);
    v8f c = zacc;
#pragma unroll
    for (int kt = 0; kt < 7; ++kt) c = wmma_f16(af[kt], frag_ld(frag, F_FNN3 + kt, lane), c);
    if (r16 < 6){
      float bv = fb3[r16];
#pragma unroll
      for (int rv = 0; rv < 8; ++rv)
        rb[(rv + 8*hf)*ROWW + RF + r16] = c[rv] + bv;
    }
  }
  __builtin_amdgcn_wave_barrier();

  // ---------------- GNN: 6 -> 36 (g_X) ----------------
#pragma unroll
  for (int nt = 0; nt < 3; ++nt){
    v8f c = zacc;
    c = wmma_f16(aX, frag_ld(frag, F_GNN + nt, lane), c);
    int n = nt*16 + r16;
    if (n < 36){
      float bv = gb[n];
#pragma unroll
      for (int rv = 0; rv < 8; ++rv)
        rb[(rv + 8*hf)*ROWW + RG + n] = c[rv] + bv;
    }
  }
  __builtin_amdgcn_wave_barrier();

  // ---------------- ICNN forward z1: 6 -> 60 ----------------
#pragma unroll
  for (int nt = 0; nt < 4; ++nt){
    v8f c = zacc;
    c = wmma_f16(aX, frag_ld(frag, F_IW1 + nt, lane), c);
    int n = nt*16 + r16;
    float bv = (n < 60) ? ib1[n] : 0.f;
#pragma unroll
    for (int rv = 0; rv < 8; ++rv){
      float z = c[rv] + bv;
      int m = rv + 8*hf;
      sg[m*SIGW + n]  = (half_t)sigmoid_f(z);
      act[m*ACTW + n] = (half_t)softplus_f(z);
    }
  }
  __builtin_amdgcn_wave_barrier();

  // ---------------- ICNN z2: 60 -> 60 (+ x skip) ----------------
  {
    v16h h0 = lds_ldA(act, lane, 0);
    v16h h1 = lds_ldA(act, lane, 32);
    __builtin_amdgcn_wave_barrier();
#pragma unroll
    for (int nt = 0; nt < 4; ++nt){
      v8f c = zacc;
      c = wmma_f16(h0, frag_ld(frag, F_IW2 + nt*2 + 0, lane), c);
      c = wmma_f16(h1, frag_ld(frag, F_IW2 + nt*2 + 1, lane), c);
      c = wmma_f16(aX, frag_ld(frag, F_IM2 + nt, lane), c);
      int n = nt*16 + r16;
      float bv = (n < 60) ? ib2[n] : 0.f;
#pragma unroll
      for (int rv = 0; rv < 8; ++rv){
        float z = c[rv] + bv;
        int m = rv + 8*hf;
        sg[m*SIGW + 64 + n] = (half_t)sigmoid_f(z);
        act[m*ACTW + n]     = (half_t)softplus_f(z);
      }
    }
  }
  __builtin_amdgcn_wave_barrier();

  // ---------------- ICNN z3: 60 -> 30 (+ x skip) ----------------
  {
    v16h h0 = lds_ldA(act, lane, 0);
    v16h h1 = lds_ldA(act, lane, 32);
    __builtin_amdgcn_wave_barrier();
#pragma unroll
    for (int nt = 0; nt < 2; ++nt){
      v8f c = zacc;
      c = wmma_f16(h0, frag_ld(frag, F_IW3 + nt*2 + 0, lane), c);
      c = wmma_f16(h1, frag_ld(frag, F_IW3 + nt*2 + 1, lane), c);
      c = wmma_f16(aX, frag_ld(frag, F_IM3 + nt, lane), c);
      int n = nt*16 + r16;
      float bv = (n < 30) ? ib3[n] : 0.f;
#pragma unroll
      for (int rv = 0; rv < 8; ++rv){
        float z = c[rv] + bv;
        int m = rv + 8*hf;
        sg[m*SIGW + 128 + n] = (half_t)sigmoid_f(z);
        act[m*ACTW + n]      = (half_t)softplus_f(z);
      }
    }
  }
  __builtin_amdgcn_wave_barrier();

  // ---------------- y = W4 h3 + b4 + I4 x (per-row VALU) ----------------
  if (lane < 16){
    const half_t* h3row = act + lane*ACTW;
    float y = ib4[0];
#pragma unroll
    for (int j = 0; j < 30; ++j) y += w4[j] * (float)h3row[j];
    const float* xr = rb + lane*ROWW + RX;
#pragma unroll
    for (int d = 0; d < 6; ++d) y += im4[d] * xr[d];
    rb[lane*ROWW + RY] = y;
  }
  __builtin_amdgcn_wave_barrier();

  // ---------------- backward: g3 = w4 * sigmoid(z3) ----------------
  if (lane < 16){
    half_t* g3row = act + lane*ACTW;
    const half_t* s3row = sg + lane*SIGW + 128;
#pragma unroll
    for (int j = 0; j < 30; ++j) g3row[j] = (half_t)(w4[j] * (float)s3row[j]);
    g3row[30] = (half_t)0.f; g3row[31] = (half_t)0.f;
  }
  __builtin_amdgcn_wave_barrier();

  v16h aG3 = lds_ldA(act, lane, 0);
  __builtin_amdgcn_wave_barrier();

  v8f cd = zacc;                                         // dy/dx accumulator (16x6 in tile 0)
  cd = wmma_f16(aG3, frag_ld(frag, F_BI3, lane), cd);    // + g3 @ I3

  // g2 = (g3 @ W3) * sigmoid(z2)
#pragma unroll
  for (int nt = 0; nt < 4; ++nt){
    v8f c = zacc;
    c = wmma_f16(aG3, frag_ld(frag, F_BW3 + nt, lane), c);
    int n = nt*16 + r16;
#pragma unroll
    for (int rv = 0; rv < 8; ++rv){
      int m = rv + 8*hf;
      act[m*ACTW + n] = (half_t)(c[rv] * (float)sg[m*SIGW + 64 + n]);
    }
  }
  __builtin_amdgcn_wave_barrier();
  v16h g2a = lds_ldA(act, lane, 0);
  v16h g2b = lds_ldA(act, lane, 32);
  __builtin_amdgcn_wave_barrier();
  cd = wmma_f16(g2a, frag_ld(frag, F_BI2 + 0, lane), cd);  // + g2 @ I2
  cd = wmma_f16(g2b, frag_ld(frag, F_BI2 + 1, lane), cd);

  // g1 = (g2 @ W2) * sigmoid(z1)
#pragma unroll
  for (int nt = 0; nt < 4; ++nt){
    v8f c = zacc;
    c = wmma_f16(g2a, frag_ld(frag, F_BW2 + nt*2 + 0, lane), c);
    c = wmma_f16(g2b, frag_ld(frag, F_BW2 + nt*2 + 1, lane), c);
    int n = nt*16 + r16;
#pragma unroll
    for (int rv = 0; rv < 8; ++rv){
      int m = rv + 8*hf;
      act[m*ACTW + n] = (half_t)(c[rv] * (float)sg[m*SIGW + n]);
    }
  }
  __builtin_amdgcn_wave_barrier();
  v16h g1a = lds_ldA(act, lane, 0);
  v16h g1b = lds_ldA(act, lane, 32);
  __builtin_amdgcn_wave_barrier();
  cd = wmma_f16(g1a, frag_ld(frag, F_BW1 + 0, lane), cd);  // + g1 @ W1
  cd = wmma_f16(g1b, frag_ld(frag, F_BW1 + 1, lane), cd);

  if (r16 < 6){
    float iv = im4[r16];                                   // + I4 row
#pragma unroll
    for (int rv = 0; rv < 8; ++rv)
      rb[(rv + 8*hf)*ROWW + RDY + r16] = cd[rv] + iv;
  }
  __builtin_amdgcn_wave_barrier();

  // ---------------- final per-row: sigma, dV, correction, output ----------------
  if (lane < 16){
    const int row = rowbase + lane;
    const float* rr = rb + lane*ROWW;
    float hh = rr[RY] - yref_p[0];
    // branchless piecewise sigma / sigma'
    float sgm = (hh >= 1.f) ? (hh - 0.5f) : ((hh > 0.f) ? 0.5f*hh*hh : 0.f);
    float sgp = (hh >= 1.f) ? 1.f         : ((hh > 0.f) ? hh          : 0.f);
    float dv[6]; float q = 0.f, sc = 0.f, nrm = 0.f;
#pragma unroll
    for (int d = 0; d < 6; ++d){
      float dxr = rr[RX + d] - xref[d];
      q += dxr * dxr;
      float t = sgp * rr[RDY + d] + 2.f * 1e-5f * dxr;
      dv[d] = t;
      sc  += t * rr[RF + d];
      nrm += t * t;
    }
    float V = sgm + 1e-5f * q;
    sc += 0.05f * V;
#pragma unroll
    for (int m = 0; m < 6; ++m){
      float t = 0.f;
#pragma unroll
      for (int d = 0; d < 6; ++d) t += dv[d] * rr[RG + d*6 + m];
      sc -= fabsf(t);
    }
    float coef = (sc > 0.f) ? (-sc / nrm) : 0.f;
    float u[6];
#pragma unroll
    for (int m = 0; m < 6; ++m) u[m] = U[row*6 + m];
#pragma unroll
    for (int d = 0; d < 6; ++d){
      float o = rr[RF + d] + coef * dv[d];
#pragma unroll
      for (int m = 0; m < 6; ++m) o += rr[RG + d*6 + m] * u[m];
      out[row*6 + d] = o;
    }
  }
}

extern "C" void kernel_launch(void* const* d_in, const int* in_sizes, int n_in,
                              void* d_out, int out_size, void* d_ws, size_t ws_size,
                              hipStream_t stream)
{
  const float* X   = (const float*)d_in[0];
  const float* U   = (const float*)d_in[1];
  const float* xr  = (const float*)d_in[2];
  const float* fw1 = (const float*)d_in[3];
  const float* fb1 = (const float*)d_in[4];
  const float* fw2 = (const float*)d_in[5];
  const float* fb2 = (const float*)d_in[6];
  const float* fw3 = (const float*)d_in[7];
  const float* fb3 = (const float*)d_in[8];
  const float* gw  = (const float*)d_in[9];
  const float* gb  = (const float*)d_in[10];
  const float* iw1 = (const float*)d_in[11];
  const float* ib1 = (const float*)d_in[12];
  const float* iw2 = (const float*)d_in[13];
  const float* ib2 = (const float*)d_in[14];
  const float* iw3 = (const float*)d_in[15];
  const float* ib3 = (const float*)d_in[16];
  const float* iw4 = (const float*)d_in[17];
  const float* ib4 = (const float*)d_in[18];
  const float* im2 = (const float*)d_in[19];
  const float* im3 = (const float*)d_in[20];
  const float* im4 = (const float*)d_in[21];

  half_t* frag = (half_t*)d_ws;
  float*  yref = (float*)((char*)d_ws + YREF_BYTE_OFF);

#define PREP(Wp, od, id, tr, foff, ks, nt) \
  holo_prepack<<<(nt)*(ks), 512, 0, stream>>>(Wp, od, id, tr, frag + (size_t)(foff)*512, ks)

  PREP(fw1,  80,   6, 1, F_FNN1, 1,  5);
  PREP(fw2, 200,  80, 1, F_FNN2, 3, 13);
  PREP(fw3,   6, 200, 1, F_FNN3, 7,  1);
  PREP(gw,   36,   6, 1, F_GNN,  1,  3);
  PREP(iw1,  60,   6, 1, F_IW1,  1,  4);
  PREP(iw2,  60,  60, 1, F_IW2,  2,  4);
  PREP(im2,  60,   6, 1, F_IM2,  1,  4);
  PREP(iw3,  30,  60, 1, F_IW3,  2,  2);
  PREP(im3,  30,   6, 1, F_IM3,  1,  2);
  PREP(iw3,  30,  60, 0, F_BW3,  1,  4);
  PREP(iw2,  60,  60, 0, F_BW2,  2,  4);
  PREP(iw1,  60,   6, 0, F_BW1,  2,  1);
  PREP(im2,  60,   6, 0, F_BI2,  2,  1);
  PREP(im3,  30,   6, 0, F_BI3,  1,  1);
#undef PREP

  holo_yref<<<1, 64, 0, stream>>>(xr, iw1, ib1, iw2, ib2, iw3, ib3, iw4, ib4,
                                  im2, im3, im4, yref);

  const int nrows  = in_sizes[0] / 6;            // 262144
  const int blocks = nrows / (16 * WAVES);       // 4096
  holo_main<<<blocks, 32 * WAVES, 0, stream>>>(X, U, xr, fb1, fb2, fb3, gb,
      ib1, ib2, ib3, ib4, iw4, im4, frag, yref, (float*)d_out);

  (void)n_in; (void)out_size; (void)ws_size;
}
